// BloomParallelAttention_65352222376478
// MI455X (gfx1250) — compile-verified, matrix operand order checked by reading, not run on hardware
//
#include <hip/hip_runtime.h>
#include <hip/hip_bf16.h>

// ---------------------------------------------------------------------------
// Bloom parallel attention for MI455X (gfx1250), bf16 WMMA + fp32 accumulate,
// async global->LDS double-buffered GEMM staging, software-pipelined A loads.
// B=2, S=2048, H=2048, NH=16, HD=128.
// ---------------------------------------------------------------------------

typedef __attribute__((ext_vector_type(16))) __bf16 v16bf;
typedef __attribute__((ext_vector_type(8)))  float  v8f;
typedef int v4i_vec __attribute__((vector_size(16)));   // matches builtin param

#define WMMA_BF16(a, b, c) \
  __builtin_amdgcn_wmma_f32_16x16x32_bf16(false, (a), false, (b), (short)0, (c), false, false)

constexpr int Bb = 2, Ss = 2048, Hh = 2048, NHc = 16, HDc = 128;
constexpr int M_ROWS = Bb * Ss;       // 4096
constexpr int N_QKV  = 3 * Hh;        // 6144
constexpr int K_DIM  = Hh;            // 2048
constexpr float INV_NORM = 0.08838834764831845f;   // 1/sqrt(128)
constexpr float NEGV = -3.402823466e38f;           // finfo(f32).min

// ---- async global->LDS support (gfx1250): feature-detected, sync fallback --
#if __has_builtin(__builtin_amdgcn_global_load_async_to_lds_b128)
#define HAVE_ASYNC_LDS 1
#else
#define HAVE_ASYNC_LDS 0
#endif

__device__ __forceinline__ void copy_b128_to_lds(const unsigned short* g,
                                                 unsigned short* l) {
#if HAVE_ASYNC_LDS
  __builtin_amdgcn_global_load_async_to_lds_b128(
      (__attribute__((address_space(1))) v4i_vec*)(const_cast<unsigned short*>(g)),
      (__attribute__((address_space(3))) v4i_vec*)(l), 0, 0);
#else
  *reinterpret_cast<uint4*>(l) = *reinterpret_cast<const uint4*>(g);
#endif
}

template <int N>
__device__ __forceinline__ void wait_async() {
#if HAVE_ASYNC_LDS
#if __has_builtin(__builtin_amdgcn_s_wait_asynccnt)
  __builtin_amdgcn_s_wait_asynccnt(N);
#else
  asm volatile("s_wait_asynccnt %0" ::"i"(N) : "memory");
#endif
#endif
}

// round-to-nearest-even fp32 -> bf16 (bit pattern as ushort)
__device__ __forceinline__ unsigned short f2bf(float x) {
  union { float f; unsigned int u; } c; c.f = x;
  unsigned int r = c.u + 0x7FFFu + ((c.u >> 16) & 1u);
  return (unsigned short)(r >> 16);
}

// Load one 16x32 bf16 A/B WMMA fragment.
// Per ISA 7.12.2 (16-bit A-matrix 16x32): lane<16 holds K 0..7 then 16..23,
// lane>=16 holds K 8..15 then 24..31, i.e. two contiguous 16B chunks at
// (row, k0 + (lane>>4)*8) and +16 elements. Caller folds row/k0 into `p`.
__device__ __forceinline__ v16bf load_frag(const unsigned short* p) {
  union { v16bf v; uint4 u[2]; } f;
  f.u[0] = *reinterpret_cast<const uint4*>(p);
  f.u[1] = *reinterpret_cast<const uint4*>(p + 16);
  return f.v;
}

// ---------------------------------------------------------------------------
__global__ void __launch_bounds__(256) cast_f32_bf16(const float* __restrict__ in,
                                                     unsigned short* __restrict__ out,
                                                     int n) {
  int i = blockIdx.x * blockDim.x + threadIdx.x;
  if (i < n) out[i] = f2bf(in[i]);
}

// ---------------------------------------------------------------------------
// Shared GEMM core: block = 256 threads (8 waves), workgroup tile 128x64,
// wave tile 16x64, K stepped by 32. Async double-buffered B staging in LDS,
// manually unrolled x2 (compile-time buffer offsets keep accumulators in
// place), A fragments software-pipelined one k-step ahead.
struct GemmCtx {
  int w, ln, hi, kofs, nblk, mblk, mrow;
};

__device__ __forceinline__ GemmCtx make_ctx(int mtile) {
  GemmCtx c;
  const int tid = threadIdx.x;
  c.w    = tid >> 5;
  const int lane = tid & 31;
  c.ln   = lane & 15;
  c.hi   = lane >> 4;
  c.kofs = c.hi * 8;
  c.nblk = blockIdx.x * 64;
  c.mblk = blockIdx.y * mtile;
  c.mrow = c.mblk + c.w * 16 + c.ln;
  return c;
}

__device__ __forceinline__ void gemm_core(const unsigned short* __restrict__ A,
                                          const unsigned short* __restrict__ W,
                                          unsigned short* bs,   // [2][64*32] LDS
                                          const GemmCtx& c, v8f acc[4]) {
  const int tid = threadIdx.x;
  const int srow = tid >> 2;            // 0..63  (B-tile row = output col)
  const int schunk = (tid & 3) * 8;     // 0,8,16,24 (k elements; 16B chunk)
  const unsigned short* gsrc = W + (size_t)(c.nblk + srow) * K_DIM + schunk;
  unsigned short* lds0 = bs + srow * 32 + schunk;
  const unsigned short* arow = A + (size_t)c.mrow * K_DIM + c.kofs;

  constexpr int NT = K_DIM / 32;        // 64 k-tiles (even)
  copy_b128_to_lds(gsrc, lds0);         // tile 0 -> buf 0
  v16bf a_cur = load_frag(arow);        // A fragment for tile 0
  v16bf a_nxt = a_cur;

  // One pipeline step: compute tile i from LDS buffer CUR, prefetch tile i+1
  // into buffer NXT (async) and its A fragment into registers.
#define GEMM_STEP(i, CUR, NXT)                                                \
  {                                                                           \
    __syncthreads();  /* all waves done reading buffer NXT (iter i-1) */      \
    if ((i) + 1 < NT) {                                                       \
      copy_b128_to_lds(gsrc + ((i) + 1) * 32, lds0 + (NXT) * (64 * 32));      \
      a_nxt = load_frag(arow + ((i) + 1) * 32);                               \
      wait_async<1>(); /* in-order completion => tile i landed (own op) */    \
    } else {                                                                  \
      wait_async<0>();                                                        \
    }                                                                         \
    __syncthreads();  /* every wave's copies of tile i landed */              \
    const unsigned short* bb = bs + (CUR) * (64 * 32);                        \
    _Pragma("unroll")                                                         \
    for (int t = 0; t < 4; ++t) {                                             \
      v16bf bf = load_frag(bb + (t * 16 + c.ln) * 32 + c.kofs);               \
      acc[t] = WMMA_BF16(a_cur, bf, acc[t]);                                  \
    }                                                                         \
    a_cur = a_nxt;                                                            \
  }

  for (int i = 0; i < NT; i += 2) {
    GEMM_STEP(i, 0, 1);
    GEMM_STEP(i + 1, 1, 0);
  }
#undef GEMM_STEP
}

// ---------------------------------------------------------------------------
// QKV GEMM: C[4096,6144] = A[M,K] * W[N,K]^T + bias, scatter to
// q[BH,S,HD], k[BH,S,HD], vT[BH,HD,S] as bf16.
__global__ void __launch_bounds__(256) qkv_gemm(const unsigned short* __restrict__ A,
                                                const unsigned short* __restrict__ W,
                                                const float* __restrict__ bias,
                                                unsigned short* __restrict__ qb,
                                                unsigned short* __restrict__ kbuf,
                                                unsigned short* __restrict__ vtb) {
  __shared__ unsigned short bs[2][64 * 32];     // 8 KB double buffer
  GemmCtx c = make_ctx(128);
  v8f acc[4] = {};
  gemm_core(A, W, &bs[0][0], c, acc);

  // Epilogue: C layout = lane n (within tile), VGPR r -> row r + 8*hi.
#pragma unroll
  for (int t = 0; t < 4; ++t) {
    int n     = c.nblk + t * 16 + c.ln;
    float bv  = bias[n];
    int head  = n / 384;            // 3*HD = 384 per head
    int rem   = n - head * 384;
    int which = rem >> 7;           // 0=q, 1=k, 2=v
    int d     = rem & 127;
#pragma unroll
    for (int r = 0; r < 8; ++r) {
      int row  = c.mblk + c.w * 16 + r + c.hi * 8;   // 0..4095
      int bidx = row >> 11;                          // /S
      int s    = row & 2047;
      int bh   = bidx * NHc + head;
      unsigned short hv = f2bf(acc[t][r] + bv);
      if (which == 0)      qb  [((size_t)bh * Ss + s) * HDc + d] = hv;
      else if (which == 1) kbuf[((size_t)bh * Ss + s) * HDc + d] = hv;
      else                 vtb [((size_t)bh * HDc + d) * Ss + s] = hv;  // transposed
    }
  }
}

// ---------------------------------------------------------------------------
// Dense GEMM: out[M,H] = ctx[M,K] * Wd[H,K]^T + b_dense + residual (fp32 out)
__global__ void __launch_bounds__(256) dense_gemm(const unsigned short* __restrict__ A,
                                                  const unsigned short* __restrict__ W,
                                                  const float* __restrict__ bias,
                                                  const float* __restrict__ residual,
                                                  float* __restrict__ out) {
  __shared__ unsigned short bs[2][64 * 32];
  GemmCtx c = make_ctx(128);
  v8f acc[4] = {};
  gemm_core(A, W, &bs[0][0], c, acc);

#pragma unroll
  for (int t = 0; t < 4; ++t) {
    int n = c.nblk + t * 16 + c.ln;
    float bv = bias[n];
#pragma unroll
    for (int r = 0; r < 8; ++r) {
      int row = c.mblk + c.w * 16 + r + c.hi * 8;
      size_t idx = (size_t)row * Hh + n;
      out[idx] = acc[t][r] + bv + residual[idx];
    }
  }
}

// ---------------------------------------------------------------------------
// Flash attention: grid (S/64, B*NH), block 128 (4 waves x 16 query rows).
__global__ void __launch_bounds__(128) attn(const unsigned short* __restrict__ qb,
                                            const unsigned short* __restrict__ kbuf,
                                            const unsigned short* __restrict__ vtb,
                                            const float* __restrict__ alibi,
                                            const unsigned char* __restrict__ mask,
                                            unsigned short* __restrict__ ctx) {
  __shared__ unsigned short lp[4 * 16 * 32];   // per-wave P tile (16x32 bf16)
  const int tid  = threadIdx.x;
  const int w    = tid >> 5;
  const int lane = tid & 31;
  const int ln   = lane & 15;
  const int hi   = lane >> 4;
  const int kofs = hi * 8;
  const int bh   = blockIdx.y;
  const int b    = bh >> 4;          // / NH
  const int head = bh & 15;
  const int q0   = blockIdx.x * 64 + w * 16;

  // Preload q fragments (A-matrix), HD=128 -> 4 fragments of K=32.
  const unsigned short* qbase = qb + ((size_t)bh * Ss + q0 + ln) * HDc;
  v16bf qf[4];
#pragma unroll
  for (int kf = 0; kf < 4; ++kf) qf[kf] = load_frag(qbase + kf * 32 + kofs);

  v8f O[8] = {};
  float mr[8], lr[8];
#pragma unroll
  for (int r = 0; r < 8; ++r) { mr[r] = NEGV; lr[r] = 0.f; }

  for (int key0 = 0; key0 < Ss; key0 += 32) {
    // ---- scores S = q * k^T : two 16x16 n-tiles of keys
    v8f s0 = {}, s1 = {};
    const unsigned short* kb0 = kbuf + ((size_t)bh * Ss + key0 + ln) * HDc + kofs;
    const unsigned short* kb1 = kb0 + (size_t)16 * HDc;
#pragma unroll
    for (int kf = 0; kf < 4; ++kf) {
      v16bf k0f = load_frag(kb0 + kf * 32);
      v16bf k1f = load_frag(kb1 + kf * 32);
      s0 = WMMA_BF16(qf[kf], k0f, s0);
      s1 = WMMA_BF16(qf[kf], k1f, s1);
    }

    // ---- alibi + scale + mask, online softmax (row r lives in one 16-lane
    //      group; shfl_xor masks 1..8 stay in-group under wave32)
    const int ka = key0 + ln, kb2 = ka + 16;
    const float al0 = alibi[(size_t)bh * Ss + ka];
    const float al1 = alibi[(size_t)bh * Ss + kb2];
    float sc[8];
#pragma unroll
    for (int r = 0; r < 8; ++r) {
      int qr = q0 + r + hi * 8;
      float v0 = s0[r] * INV_NORM + al0;
      float v1 = s1[r] * INV_NORM + al1;
      if (mask[(size_t)qr * Ss + ka])  v0 = NEGV;
      if (mask[(size_t)qr * Ss + kb2]) v1 = NEGV;

      float mx = fmaxf(v0, v1);
      mx = fmaxf(mx, __shfl_xor(mx, 1));
      mx = fmaxf(mx, __shfl_xor(mx, 2));
      mx = fmaxf(mx, __shfl_xor(mx, 4));
      mx = fmaxf(mx, __shfl_xor(mx, 8));
      float mnew = fmaxf(mr[r], mx);

      float p0 = __expf(v0 - mnew);
      float p1 = __expf(v1 - mnew);
      float ps = p0 + p1;
      ps += __shfl_xor(ps, 1);
      ps += __shfl_xor(ps, 2);
      ps += __shfl_xor(ps, 4);
      ps += __shfl_xor(ps, 8);

      float f = __expf(mr[r] - mnew);
      lr[r] = lr[r] * f + ps;
      mr[r] = mnew;
      sc[r] = f;

      int prow = r + hi * 8;
      lp[(w * 16 + prow) * 32 + ln]      = f2bf(p0);
      lp[(w * 16 + prow) * 32 + ln + 16] = f2bf(p1);
    }

    // rescale running context
#pragma unroll
    for (int dt = 0; dt < 8; ++dt)
#pragma unroll
      for (int r = 0; r < 8; ++r)
        O[dt][r] *= sc[r];

    // ---- O += P * V : P read back from LDS in A-fragment layout,
    //      V fragments contiguous thanks to the transposed vT buffer.
    v16bf pf = load_frag(&lp[(w * 16 + ln) * 32 + kofs]);
    const unsigned short* vbase = vtb + ((size_t)bh * HDc + ln) * Ss + key0 + kofs;
#pragma unroll
    for (int dt = 0; dt < 8; ++dt) {
      v16bf vf = load_frag(vbase + (size_t)dt * 16 * Ss);
      O[dt] = WMMA_BF16(pf, vf, O[dt]);
    }
  }

  // ---- epilogue: ctx[b, s, head*HD + d] = O / l  (bf16 for dense GEMM)
#pragma unroll
  for (int r = 0; r < 8; ++r) {
    int qr = q0 + r + hi * 8;
    float inv_l = 1.0f / lr[r];
    size_t rowbase = ((size_t)(b * Ss + qr)) * Hh + head * HDc;
#pragma unroll
    for (int dt = 0; dt < 8; ++dt)
      ctx[rowbase + dt * 16 + ln] = f2bf(O[dt][r] * inv_l);
  }
}

// ---------------------------------------------------------------------------
extern "C" void kernel_launch(void* const* d_in, const int* in_sizes, int n_in,
                              void* d_out, int out_size, void* d_ws, size_t ws_size,
                              hipStream_t stream) {
  const float*         hidden   = (const float*)d_in[0];
  const float*         residual = (const float*)d_in[1];
  const float*         alibi    = (const float*)d_in[2];
  const unsigned char* mask     = (const unsigned char*)d_in[3];  // jax bool = 1 byte
  const float*         w_qkv    = (const float*)d_in[4];
  const float*         b_qkv    = (const float*)d_in[5];
  const float*         w_dense  = (const float*)d_in[6];
  const float*         b_dense  = (const float*)d_in[7];
  float*               out      = (float*)d_out;

  // workspace carve-up (bf16 elements) -- total ~117 MB
  unsigned short* ws = (unsigned short*)d_ws;
  size_t o = 0;
  unsigned short* hid_bf  = ws + o; o += (size_t)M_ROWS * Hh;       // 8.39M
  unsigned short* wqkv_bf = ws + o; o += (size_t)N_QKV * Hh;        // 12.58M
  unsigned short* wd_bf   = ws + o; o += (size_t)Hh * Hh;           // 4.19M
  const size_t qkv_elems  = (size_t)Bb * NHc * Ss * HDc;            // 8.39M each
  unsigned short* qb      = ws + o; o += qkv_elems;
  unsigned short* kb      = ws + o; o += qkv_elems;
  unsigned short* vtb     = ws + o; o += qkv_elems;
  unsigned short* ctx     = ws + o; o += (size_t)M_ROWS * Hh;

  const int nh = M_ROWS * Hh;      // 8,388,608
  const int nq = N_QKV * Hh;       // 12,582,912
  const int nd = Hh * Hh;          // 4,194,304
  cast_f32_bf16<<<(nh + 255) / 256, 256, 0, stream>>>(hidden, hid_bf, nh);
  cast_f32_bf16<<<(nq + 255) / 256, 256, 0, stream>>>(w_qkv, wqkv_bf, nq);
  cast_f32_bf16<<<(nd + 255) / 256, 256, 0, stream>>>(w_dense, wd_bf, nd);

  qkv_gemm<<<dim3(N_QKV / 64, M_ROWS / 128), 256, 0, stream>>>(
      hid_bf, wqkv_bf, b_qkv, qb, kb, vtb);

  attn<<<dim3(Ss / 64, Bb * NHc), 128, 0, stream>>>(
      qb, kb, vtb, alibi, mask, ctx);

  dense_gemm<<<dim3(Hh / 64, M_ROWS / 128), 256, 0, stream>>>(
      ctx, wd_bf, b_dense, residual, out);
}
